// CausalSelfAttention_77773267796123
// MI455X (gfx1250) — compile-verified, matrix operand order checked by reading, not run on hardware
//
#include <hip/hip_runtime.h>
#include <hip/hip_bf16.h>

#define D_MODEL 1024
#define N_HEAD  16
#define DH      64
#define SEQ     2048
#define BATCH   2
#define NTOK    (BATCH * SEQ)   // 4096

typedef __bf16 bf16_t;
typedef bf16_t v16bf __attribute__((ext_vector_type(16)));
typedef bf16_t v8bf  __attribute__((ext_vector_type(8)));
typedef float  v8f   __attribute__((ext_vector_type(8)));
typedef float  v4f   __attribute__((ext_vector_type(4)));

union Frag16 { v16bf v; v8bf h[2]; };

// A/B fragment loader for 16x16x32 bf16 WMMA when the contraction dim is
// contiguous at `row`:  lane (m or n) = tid&15, half = tid>>4.
// elements 0..7  <- row[8*half .. 8*half+7]
// elements 8..15 <- row[16+8*half .. 16+8*half+7]
__device__ __forceinline__ v16bf load_frag(const bf16_t* row, int half) {
  Frag16 f;
  f.h[0] = *(const v8bf*)(row + 8 * half);
  f.h[1] = *(const v8bf*)(row + 16 + 8 * half);
  return f.v;
}

__device__ __forceinline__ v8f wmma_bf16(v16bf a, v16bf b, v8f c) {
  return __builtin_amdgcn_wmma_f32_16x16x32_bf16(false, a, false, b, (short)0, c,
                                                 false, false);
}

// ---------------- precision / layout conversion ----------------

__global__ void cvt_f32_bf16(const float* __restrict__ src,
                             bf16_t* __restrict__ dst, int n) {
  int i = blockIdx.x * blockDim.x + threadIdx.x;
  if (i < n) dst[i] = (bf16_t)src[i];
}

// W: K x N row-major (fp32)  ->  Wt: N x K row-major (bf16)
__global__ void transpose_to_bf16(const float* __restrict__ W,
                                  bf16_t* __restrict__ Wt, int K, int N) {
  long i = (long)blockIdx.x * blockDim.x + threadIdx.x;
  long total = (long)K * N;
  if (i < total) {
    int n = (int)(i / K), c = (int)(i % K);
    Wt[i] = (bf16_t)W[(long)c * N + n];
  }
}

// ---------------- projection: kq = x @ W_kq + b_kq ----------------
// outT orientation: A = Wkqt rows (features, M), B = x rows (tokens, N).
// 64 features x 64 tokens per wave: 16 C tiles, 16 WMMAs per K-step.
// C tile layout: lane = token (N), VGPR j = feature (M = j + 8*half).
__global__ __launch_bounds__(32) void proj_kq_kernel(
    const bf16_t* __restrict__ xb, const bf16_t* __restrict__ Wkqt,
    const float* __restrict__ bias, bf16_t* __restrict__ kbuf,
    bf16_t* __restrict__ qbuf) {
  const int lane = threadIdx.x;
  const int n16 = lane & 15, half = lane >> 4;
  const int t0 = blockIdx.x * 64;   // token group (4 tiles)
  const int f0 = blockIdx.y * 64;   // feature group of 2*D (4 tiles)
  v8f acc[4][4] = {};               // [feature tile][token tile]
  for (int c = 0; c < D_MODEL; c += 32) {
    v16bf bfrag[4];
#pragma unroll
    for (int u = 0; u < 4; ++u)
      bfrag[u] = load_frag(xb + (long)(t0 + u * 16 + n16) * D_MODEL + c, half);
#pragma unroll
    for (int t = 0; t < 4; ++t) {
      v16bf afrag =
          load_frag(Wkqt + (long)(f0 + t * 16 + n16) * D_MODEL + c, half);
#pragma unroll
      for (int u = 0; u < 4; ++u)
        acc[t][u] = wmma_bf16(afrag, bfrag[u], acc[t][u]);
    }
  }
#pragma unroll
  for (int t = 0; t < 4; ++t) {
    const int f8 = f0 + t * 16 + 8 * half;  // 8 consecutive features
    float bia[8];
#pragma unroll
    for (int j = 0; j < 8; ++j) bia[j] = bias[f8 + j];
#pragma unroll
    for (int u = 0; u < 4; ++u) {
      const int tok = t0 + u * 16 + n16;
      const int b = tok / SEQ, l = tok % SEQ;
      v8bf outv;
      if (f8 < D_MODEL) {  // K part
#pragma unroll
        for (int j = 0; j < 8; ++j) outv[j] = (bf16_t)(acc[t][u][j] + bia[j]);
        const int h = f8 / DH, d0 = f8 % DH;
        *(v8bf*)(kbuf + (((long)(b * N_HEAD + h) * SEQ + l) * DH + d0)) = outv;
      } else {             // Q part, fold in softmax scale 1/sqrt(64)
        const int fq = f8 - D_MODEL;
#pragma unroll
        for (int j = 0; j < 8; ++j)
          outv[j] = (bf16_t)((acc[t][u][j] + bia[j]) * 0.125f);
        const int h = fq / DH, d0 = fq % DH;
        *(v8bf*)(qbuf + (((long)(b * N_HEAD + h) * SEQ + l) * DH + d0)) = outv;
      }
    }
  }
}

// ---------------- projection: v = x @ W_v + b_v, stored transposed ----------------
// A = x rows (tokens, M), B = Wvt rows (features, N). 64 tokens x 64 features.
// C tile layout: lane = feature, VGPR j = token -> contiguous store into
// Vt (b,h,dh,L).
__global__ __launch_bounds__(32) void proj_v_kernel(
    const bf16_t* __restrict__ xb, const bf16_t* __restrict__ Wvt,
    const float* __restrict__ bias, bf16_t* __restrict__ vtb) {
  const int lane = threadIdx.x;
  const int n16 = lane & 15, half = lane >> 4;
  const int t0 = blockIdx.x * 64;  // token group (4 tiles)
  const int f0 = blockIdx.y * 64;  // feature group (4 tiles)
  v8f acc[4][4] = {};              // [token tile][feature tile]
  for (int c = 0; c < D_MODEL; c += 32) {
    v16bf wfrag[4];
#pragma unroll
    for (int u = 0; u < 4; ++u)
      wfrag[u] = load_frag(Wvt + (long)(f0 + u * 16 + n16) * D_MODEL + c, half);
#pragma unroll
    for (int t = 0; t < 4; ++t) {
      v16bf xfrag =
          load_frag(xb + (long)(t0 + t * 16 + n16) * D_MODEL + c, half);
#pragma unroll
      for (int u = 0; u < 4; ++u)
        acc[t][u] = wmma_bf16(xfrag, wfrag[u], acc[t][u]);
    }
  }
#pragma unroll
  for (int u = 0; u < 4; ++u) {
    const int feat = f0 + u * 16 + n16;
    const int h = feat / DH, d = feat % DH;
    const float bv = bias[feat];
#pragma unroll
    for (int t = 0; t < 4; ++t) {
      const int tok8 = t0 + t * 16 + 8 * half;  // 8 consecutive tokens, same b
      const int b = tok8 / SEQ, l0 = tok8 % SEQ;
      v8bf outv;
#pragma unroll
      for (int j = 0; j < 8; ++j) outv[j] = (bf16_t)(acc[t][u][j] + bv);
      *(v8bf*)(vtb + (((long)(b * N_HEAD + h) * DH + d) * SEQ + l0)) = outv;
    }
  }
}

// ---------------- flash attention, one wave per 32-query tile ----------------
// Computes S^T = K @ Q^T (A = K rows, B = Q rows); two query tiles per wave
// share all K and V fragment loads. In C-layout each lane owns one query
// column per tile -> softmax stats are per-lane + one shfl_xor(16).
// The softmaxed P tile in C-layout is exactly the B operand of O^T = V^T @ P^T.
__global__ __launch_bounds__(32) void attn_kernel(
    const bf16_t* __restrict__ qbuf, const bf16_t* __restrict__ kbuf,
    const bf16_t* __restrict__ vtb, bf16_t* __restrict__ attnb) {
  const int lane = threadIdx.x;
  const int n16 = lane & 15, half = lane >> 4;
  const int q0 = blockIdx.x * 32;
  const int bh = blockIdx.y;       // b*H + h
  int qg[2];
  v16bf qf[2][2];
  v8f o[2][4] = {};
  float m_i[2], l_i[2];
#pragma unroll
  for (int g = 0; g < 2; ++g) {
    qg[g] = q0 + g * 16 + n16;     // this lane's query for tile g
    const bf16_t* qrow = qbuf + ((long)bh * SEQ + qg[g]) * DH;
    qf[g][0] = load_frag(qrow, half);       // c = 0..31
    qf[g][1] = load_frag(qrow + 32, half);  // c = 32..63
    m_i[g] = -3.0e38f;
    l_i[g] = 0.0f;
  }
  const int nkb = (q0 + 63) / 32;  // 32-key blocks needed (causal)
  for (int kb = 0; kb < nkb; ++kb) {
    const int k0 = kb * 32;
    v8f s[2][2];                   // [query tile][key tile]
#pragma unroll
    for (int t = 0; t < 2; ++t) {
      const bf16_t* krow = kbuf + ((long)bh * SEQ + k0 + t * 16 + n16) * DH;
      v16bf a0 = load_frag(krow, half);
      v16bf a1 = load_frag(krow + 32, half);
#pragma unroll
      for (int g = 0; g < 2; ++g) {
        v8f c = {};
        c = wmma_bf16(a0, qf[g][0], c);
        c = wmma_bf16(a1, qf[g][1], c);
        s[g][t] = c;
      }
    }
    v16bf pfrag[2];
#pragma unroll
    for (int g = 0; g < 2; ++g) {
      // causal mask + running max
      float mx = m_i[g];
#pragma unroll
      for (int t = 0; t < 2; ++t)
#pragma unroll
        for (int j = 0; j < 8; ++j) {
          const int key = k0 + t * 16 + j + 8 * half;
          const float v = (key <= qg[g]) ? s[g][t][j] : -3.0e38f;
          s[g][t][j] = v;
          mx = fmaxf(mx, v);
        }
      mx = fmaxf(mx, __shfl_xor(mx, 16, 32));
      const float alpha = __expf(m_i[g] - mx);
      m_i[g] = mx;
      float rs = 0.0f;
#pragma unroll
      for (int t = 0; t < 2; ++t)
#pragma unroll
        for (int j = 0; j < 8; ++j) {
          const float p = __expf(s[g][t][j] - mx);
          s[g][t][j] = p;
          rs += p;
        }
      rs += __shfl_xor(rs, 16, 32);
      l_i[g] = l_i[g] * alpha + rs;
      // P^T as B operand: element i<8 -> tile0[i], i>=8 -> tile1[i-8]
#pragma unroll
      for (int j = 0; j < 8; ++j) {
        pfrag[g][j]     = (bf16_t)s[g][0][j];
        pfrag[g][8 + j] = (bf16_t)s[g][1][j];
      }
#pragma unroll
      for (int dc = 0; dc < 4; ++dc)
#pragma unroll
        for (int j = 0; j < 8; ++j) o[g][dc][j] *= alpha;
    }
#pragma unroll
    for (int dc = 0; dc < 4; ++dc) {
      const bf16_t* vrow = vtb + ((long)bh * DH + dc * 16 + n16) * SEQ + k0;
      v16bf va = load_frag(vrow, half);
#pragma unroll
      for (int g = 0; g < 2; ++g) o[g][dc] = wmma_bf16(va, pfrag[g], o[g][dc]);
    }
  }
  const int b = bh / N_HEAD, h = bh % N_HEAD;
#pragma unroll
  for (int g = 0; g < 2; ++g) {
    const float inv = 1.0f / l_i[g];
    bf16_t* orow = attnb + ((long)b * SEQ + qg[g]) * D_MODEL + h * DH;
#pragma unroll
    for (int dc = 0; dc < 4; ++dc) {
      v8bf outv;
#pragma unroll
      for (int j = 0; j < 8; ++j) outv[j] = (bf16_t)(o[g][dc][j] * inv);
      *(v8bf*)(orow + dc * 16 + 8 * half) = outv;
    }
  }
}

// ---------------- output projection: out = attn @ W_o + b_o (fp32) ----------------
// A = Wot rows (features, M), B = attn rows (tokens, N). 64x64 per wave.
__global__ __launch_bounds__(32) void proj_o_kernel(
    const bf16_t* __restrict__ attnb, const bf16_t* __restrict__ Wot,
    const float* __restrict__ bias, float* __restrict__ out) {
  const int lane = threadIdx.x;
  const int n16 = lane & 15, half = lane >> 4;
  const int t0 = blockIdx.x * 64;
  const int f0 = blockIdx.y * 64;
  v8f acc[4][4] = {};  // [feature tile][token tile]
  for (int c = 0; c < D_MODEL; c += 32) {
    v16bf bfrag[4];
#pragma unroll
    for (int u = 0; u < 4; ++u)
      bfrag[u] =
          load_frag(attnb + (long)(t0 + u * 16 + n16) * D_MODEL + c, half);
#pragma unroll
    for (int t = 0; t < 4; ++t) {
      v16bf afrag =
          load_frag(Wot + (long)(f0 + t * 16 + n16) * D_MODEL + c, half);
#pragma unroll
      for (int u = 0; u < 4; ++u)
        acc[t][u] = wmma_bf16(afrag, bfrag[u], acc[t][u]);
    }
  }
#pragma unroll
  for (int t = 0; t < 4; ++t) {
    const int f8 = f0 + t * 16 + 8 * half;
    float bia[8];
#pragma unroll
    for (int j = 0; j < 8; ++j) bia[j] = bias[f8 + j];
#pragma unroll
    for (int u = 0; u < 4; ++u) {
      const int tok = t0 + u * 16 + n16;
      float* outrow = out + (long)tok * D_MODEL;
      v4f lo, hi;
#pragma unroll
      for (int j = 0; j < 4; ++j) {
        lo[j] = acc[t][u][j] + bia[j];
        hi[j] = acc[t][u][4 + j] + bia[4 + j];
      }
      *(v4f*)(outrow + f8) = lo;
      *(v4f*)(outrow + f8 + 4) = hi;
    }
  }
}

extern "C" void kernel_launch(void* const* d_in, const int* in_sizes, int n_in,
                              void* d_out, int out_size, void* d_ws,
                              size_t ws_size, hipStream_t stream) {
  const float* x   = (const float*)d_in[0];
  const float* Wkq = (const float*)d_in[1];
  const float* bkq = (const float*)d_in[2];
  const float* Wv  = (const float*)d_in[3];
  const float* bv  = (const float*)d_in[4];
  const float* Wo  = (const float*)d_in[5];
  const float* bo  = (const float*)d_in[6];
  float* out = (float*)d_out;

  char* ws = (char*)d_ws;
  bf16_t* xb    = (bf16_t*)(ws + 0);          //  8 MB  x in bf16
  bf16_t* Wkqt  = (bf16_t*)(ws + 8388608);    //  4 MB  W_kq^T
  bf16_t* Wvt   = (bf16_t*)(ws + 12582912);   //  2 MB  W_v^T
  bf16_t* Wot   = (bf16_t*)(ws + 14680064);   //  2 MB  W_o^T
  bf16_t* kbuf  = (bf16_t*)(ws + 16777216);   //  8 MB  K (b,h,l,dh)
  bf16_t* qbuf  = (bf16_t*)(ws + 25165824);   //  8 MB  Q (b,h,l,dh), pre-scaled
  bf16_t* vtb   = (bf16_t*)(ws + 33554432);   //  8 MB  V^T (b,h,dh,l)
  bf16_t* attnb = (bf16_t*)(ws + 41943040);   //  8 MB  attention out (b*l, D)

  cvt_f32_bf16<<<(NTOK * D_MODEL + 255) / 256, 256, 0, stream>>>(
      x, xb, NTOK * D_MODEL);
  transpose_to_bf16<<<(2 * D_MODEL * D_MODEL + 255) / 256, 256, 0, stream>>>(
      Wkq, Wkqt, D_MODEL, 2 * D_MODEL);
  transpose_to_bf16<<<(D_MODEL * D_MODEL + 255) / 256, 256, 0, stream>>>(
      Wv, Wvt, D_MODEL, D_MODEL);
  transpose_to_bf16<<<(D_MODEL * D_MODEL + 255) / 256, 256, 0, stream>>>(
      Wo, Wot, D_MODEL, D_MODEL);

  proj_kq_kernel<<<dim3(NTOK / 64, 2 * D_MODEL / 64), 32, 0, stream>>>(
      xb, Wkqt, bkq, kbuf, qbuf);
  proj_v_kernel<<<dim3(NTOK / 64, D_MODEL / 64), 32, 0, stream>>>(
      xb, Wvt, bv, vtb);
  attn_kernel<<<dim3(SEQ / 32, BATCH * N_HEAD), 32, 0, stream>>>(
      qbuf, kbuf, vtb, attnb);
  proj_o_kernel<<<dim3(NTOK / 64, D_MODEL / 64), 32, 0, stream>>>(
      attnb, Wot, bo, out);
}
